// WanSelfAttention_15599321219097
// MI455X (gfx1250) — compile-verified, matrix operand order checked by reading, not run on hardware
//
#include <hip/hip_runtime.h>
#include <hip/hip_bf16.h>

typedef unsigned short u16;
typedef __attribute__((ext_vector_type(16))) __bf16 v16bf;
typedef __attribute__((ext_vector_type(8)))  float  v8f;

#define DIM 1536
#define NHEADS 12
#define HDIM 128
#define BATCH 2
#define SEQ 3072
#define MROWS (BATCH*SEQ)

#if __has_builtin(__builtin_amdgcn_global_load_async_to_lds_b128) && \
    __has_builtin(__builtin_amdgcn_s_wait_asynccnt)
#define HAS_ASYNC 1
#else
#define HAS_ASYNC 0
#endif

// ---------- helpers ----------
__device__ __forceinline__ u16 f2bf(float f) {
  unsigned int u = __float_as_uint(f);
  u += 0x7fffu + ((u >> 16) & 1u);       // round-to-nearest-even
  return (u16)(u >> 16);
}

__device__ __forceinline__ v16bf load_frag(const u16* p0, const u16* p1) {
  union { v16bf v; uint4 q[2]; } u;
  u.q[0] = *(const uint4*)p0;
  u.q[1] = *(const uint4*)p1;
  return u.v;
}

__device__ __forceinline__ v8f wmma_bf16(v16bf a, v16bf b, v8f c) {
  return __builtin_amdgcn_wmma_f32_16x16x32_bf16(false, a, false, b,
                                                 (short)0, c, false, false);
}

#if HAS_ASYNC
typedef int v4i_vs __attribute__((vector_size(16)));
typedef __attribute__((address_space(1))) v4i_vs* gas_v4i;
typedef __attribute__((address_space(3))) v4i_vs* las_v4i;
__device__ __forceinline__ void async_cp16(const void* g, void* l) {
  __builtin_amdgcn_global_load_async_to_lds_b128((gas_v4i)g, (las_v4i)l, 0, 0);
}
#endif

// ---------- prep: fp32 -> bf16 (vectorized) ----------
__global__ __launch_bounds__(256) void cvt_bf16(const float* __restrict__ x,
                                                u16* __restrict__ y) {
  const int i = (blockIdx.x * 256 + threadIdx.x) * 4;
  float4 f = *(const float4*)(x + i);
  y[i + 0] = f2bf(f.x); y[i + 1] = f2bf(f.y);
  y[i + 2] = f2bf(f.z); y[i + 3] = f2bf(f.w);
}

// ---------- prep: W(KxN) fp32 -> Wt(NxK) bf16 ----------
__global__ __launch_bounds__(256) void transpose_bf16(
    const float* __restrict__ W, u16* __restrict__ Wt, int K, int N)
{
  __shared__ float tile[32][33];
  const int bn = blockIdx.x * 32;              // n tile
  const int bk = blockIdx.y * 32;              // k tile
  const int tx = threadIdx.x & 31, ty = threadIdx.x >> 5;
  #pragma unroll
  for (int i = 0; i < 32; i += 8)
    tile[ty + i][tx] = W[(size_t)(bk + ty + i) * N + bn + tx];
  __syncthreads();
  #pragma unroll
  for (int i = 0; i < 32; i += 8)
    Wt[(size_t)(bn + ty + i) * K + bk + tx] = f2bf(tile[tx][ty + i]);
}

// ---------- kernel 1: C = A(MxK,bf16) * Wt(NxK,bf16)^T + bias, fp32 out ----------
__global__ __launch_bounds__(256) void gemm_bf16(
    const u16* __restrict__ A, const u16* __restrict__ Wt,
    const float* __restrict__ bias, float* __restrict__ C,
    int M, int K, int N)
{
  __shared__ alignas(16) u16 Asub[2][128][40];   // double-buffered, padded pitch
  __shared__ alignas(16) u16 Bsub[2][128][40];

  const int t    = threadIdx.x;
  const int wave = t >> 5, lane = t & 31;
  const int half = lane >> 4, lr = lane & 15;
  const int wm   = wave & 3;
  const int wn   = wave >> 2;
  const int mBase = blockIdx.y * 128;
  const int nBase = blockIdx.x * 128;

  // staging role: 2 threads per row, 16-half (32B) segment each
  const int sr = t >> 1;
  const int sc = (t & 1) * 16;
  const u16* aRow = A  + (size_t)(mBase + sr) * K + sc;
  const u16* bRow = Wt + (size_t)(nBase + sr) * K + sc;

  v8f acc[2][4];
  #pragma unroll
  for (int i = 0; i < 2; ++i)
    #pragma unroll
    for (int j = 0; j < 4; ++j) acc[i][j] = (v8f)0.0f;

  auto stage = [&](int buf, int kb) {
#if HAS_ASYNC
    async_cp16(aRow + kb,     &Asub[buf][sr][sc]);
    async_cp16(aRow + kb + 8, &Asub[buf][sr][sc + 8]);
    async_cp16(bRow + kb,     &Bsub[buf][sr][sc]);
    async_cp16(bRow + kb + 8, &Bsub[buf][sr][sc + 8]);
#else
    *(uint4*)&Asub[buf][sr][sc]     = *(const uint4*)(aRow + kb);
    *(uint4*)&Asub[buf][sr][sc + 8] = *(const uint4*)(aRow + kb + 8);
    *(uint4*)&Bsub[buf][sr][sc]     = *(const uint4*)(bRow + kb);
    *(uint4*)&Bsub[buf][sr][sc + 8] = *(const uint4*)(bRow + kb + 8);
#endif
  };

  stage(0, 0);
  int i = 0;
  for (int kb = 0; kb < K; kb += 32, ++i) {
    const int  cur  = i & 1;
    const bool more = (kb + 32) < K;
    if (more) stage(cur ^ 1, kb + 32);           // prefetch next tile
#if HAS_ASYNC
    if (more) __builtin_amdgcn_s_wait_asynccnt(4);   // current tile landed
    else      __builtin_amdgcn_s_wait_asynccnt(0);
#endif
    __syncthreads();

    v16bf af[2], bfr[4];
    #pragma unroll
    for (int mi = 0; mi < 2; ++mi) {
      const u16* r0 = &Asub[cur][32*wm + 16*mi + lr][8*half];
      af[mi] = load_frag(r0, r0 + 16);
    }
    #pragma unroll
    for (int ni = 0; ni < 4; ++ni) {
      const u16* r0 = &Bsub[cur][64*wn + 16*ni + lr][8*half];
      bfr[ni] = load_frag(r0, r0 + 16);
    }
    #pragma unroll
    for (int mi = 0; mi < 2; ++mi)
      #pragma unroll
      for (int ni = 0; ni < 4; ++ni)
        acc[mi][ni] = wmma_bf16(af[mi], bfr[ni], acc[mi][ni]);
    __syncthreads();
  }

  #pragma unroll
  for (int mi = 0; mi < 2; ++mi)
    #pragma unroll
    for (int ni = 0; ni < 4; ++ni) {
      const int n = nBase + 64*wn + 16*ni + lr;
      const float bv = bias[n];
      #pragma unroll
      for (int r = 0; r < 8; ++r) {
        const int m = mBase + 32*wm + 16*mi + r + 8*half;
        C[(size_t)m * N + n] = acc[mi][ni][r] + bv;
      }
    }
}

// ---------- kernel 2: RMSNorm + RoPE (q,k) / bf16 convert (v) ----------
__global__ __launch_bounds__(256) void norm_rope(
    const float* __restrict__ qf, const float* __restrict__ kf,
    const float* __restrict__ vf,
    const float* __restrict__ gq, const float* __restrict__ gk,
    const int* __restrict__ grid_sizes, const float* __restrict__ freqs,
    u16* __restrict__ qb, u16* __restrict__ kb, u16* __restrict__ vb, int L)
{
  const int R     = blockIdx.x;
  const int which = blockIdx.y;
  const int b = R / L, l = R % L;
  const int t = threadIdx.x;

  const float* src; const float* g; u16* dst;
  if (which == 0)      { src = qf; g = gq; dst = qb; }
  else if (which == 1) { src = kf; g = gk; dst = kb; }
  else                 { src = vf; g = nullptr; dst = vb; }

  const float* row  = src + (size_t)R * DIM;
  u16*         orow = dst + (size_t)R * DIM;

  if (which == 2) {
    for (int i = t; i < DIM; i += 256) orow[i] = f2bf(row[i]);
    return;
  }

  float s = 0.f;
  #pragma unroll
  for (int i = 0; i < 6; ++i) { float v = row[t*6 + i]; s += v * v; }
  __shared__ float red[256];
  red[t] = s;
  __syncthreads();
  for (int off = 128; off > 0; off >>= 1) {
    if (t < off) red[t] += red[t + off];
    __syncthreads();
  }
  const float factor = rsqrtf(red[0] * (1.0f / DIM) + 1e-6f);

  const int f = grid_sizes[b*3+0], h = grid_sizes[b*3+1], w = grid_sizes[b*3+2];
  const int hw = h * w;
  const int fi = l / hw, hi = (l / w) % h, wi = l % w;
  const bool valid = l < f * hw;

  #pragma unroll
  for (int p = 0; p < 3; ++p) {
    const int pg = t*3 + p;
    const int j  = pg & 63;
    const int e0 = 2 * pg;
    const float xr  = row[e0]     * factor * g[e0];
    const float xim = row[e0 + 1] * factor * g[e0 + 1];
    float orr, oii;
    if (valid) {
      const int fr = (j < 22) ? fi : ((j < 43) ? hi : wi);
      const float ang = freqs[fr * 64 + j];
      const float cs = __cosf(ang), sn = __sinf(ang);
      orr = xr * cs - xim * sn;
      oii = xr * sn + xim * cs;
    } else { orr = xr; oii = xim; }
    orow[e0]     = f2bf(orr);
    orow[e0 + 1] = f2bf(oii);
  }
}

// ---------- kernel 3: flash attention (bf16 in, bf16 out) ----------
__global__ __launch_bounds__(256) void attn(
    const u16* __restrict__ qb, const u16* __restrict__ kb,
    const u16* __restrict__ vb, const int* __restrict__ seq_lens,
    u16* __restrict__ out, int L)
{
  __shared__ alignas(16) u16 Kt[32][128];
  __shared__ alignas(16) u16 Vt[128][32];
  __shared__ alignas(16) u16 Pl[8][16][32];

  const int b = blockIdx.z, h = blockIdx.y;
  const int t = threadIdx.x, wave = t >> 5, lane = t & 31;
  const int half = lane >> 4, lr = lane & 15;
  const int seqlen = seq_lens[b];
  const float scale = 0.08838834764831845f;

  const int qrow0 = blockIdx.x * 128 + wave * 16;
  v16bf qfr[4];
  {
    const u16* qp = qb + ((size_t)(b*L + qrow0 + lr) * NHEADS + h) * HDIM;
    #pragma unroll
    for (int c = 0; c < 4; ++c) {
      const u16* p0 = qp + 32*c + 8*half;
      qfr[c] = load_frag(p0, p0 + 16);
    }
  }

  v8f o[8];
  #pragma unroll
  for (int j = 0; j < 8; ++j) o[j] = (v8f)0.0f;
  float rM[8], rL[8];
  #pragma unroll
  for (int r = 0; r < 8; ++r) { rM[r] = -1e30f; rL[r] = 0.f; }

  const int key = t >> 3;
  const int ds0 = (t & 7) * 16;

  for (int kt = 0; kt < L; kt += 32) {
    {
      const u16* ksrc = kb + ((size_t)(b*L + kt + key) * NHEADS + h) * HDIM + ds0;
#if HAS_ASYNC
      async_cp16(ksrc,     &Kt[key][ds0]);
      async_cp16(ksrc + 8, &Kt[key][ds0 + 8]);
#else
      *(uint4*)&Kt[key][ds0]     = *(const uint4*)ksrc;
      *(uint4*)&Kt[key][ds0 + 8] = *(const uint4*)(ksrc + 8);
#endif
      const u16* vsrc = vb + ((size_t)(b*L + kt + key) * NHEADS + h) * HDIM + ds0;
      #pragma unroll
      for (int i = 0; i < 16; ++i) Vt[ds0 + i][key] = vsrc[i];
    }
#if HAS_ASYNC
    __builtin_amdgcn_s_wait_asynccnt(0);
#endif
    __syncthreads();

    v8f s0 = (v8f)0.0f, s1 = (v8f)0.0f;
    #pragma unroll
    for (int c = 0; c < 4; ++c) {
      const u16* k0 = &Kt[lr][32*c + 8*half];
      const u16* k1 = &Kt[16 + lr][32*c + 8*half];
      s0 = wmma_bf16(qfr[c], load_frag(k0, k0 + 16), s0);
      s1 = wmma_bf16(qfr[c], load_frag(k1, k1 + 16), s1);
    }

    const bool v0 = (kt + lr)      < seqlen;
    const bool v1 = (kt + 16 + lr) < seqlen;

    #pragma unroll
    for (int r = 0; r < 8; ++r) {
      float a0 = v0 ? s0[r] * scale : -1e30f;
      float a1 = v1 ? s1[r] * scale : -1e30f;
      float tmax = fmaxf(a0, a1);
      #pragma unroll
      for (int x = 1; x < 16; x <<= 1)
        tmax = fmaxf(tmax, __shfl_xor(tmax, x, 16));
      const float nM  = fmaxf(rM[r], tmax);
      const float fac = __expf(rM[r] - nM);
      const float p0 = v0 ? __expf(a0 - nM) : 0.f;
      const float p1 = v1 ? __expf(a1 - nM) : 0.f;
      float ps = p0 + p1;
      #pragma unroll
      for (int x = 1; x < 16; x <<= 1) ps += __shfl_xor(ps, x, 16);
      rL[r] = rL[r] * fac + ps;
      rM[r] = nM;
      #pragma unroll
      for (int j = 0; j < 8; ++j) o[j][r] *= fac;
      Pl[wave][r + 8*half][lr]      = f2bf(p0);
      Pl[wave][r + 8*half][16 + lr] = f2bf(p1);
    }

    const u16* pp = &Pl[wave][lr][8*half];
    const v16bf pf = load_frag(pp, pp + 16);
    #pragma unroll
    for (int j = 0; j < 8; ++j) {
      const u16* vp = &Vt[16*j + lr][8*half];
      o[j] = wmma_bf16(pf, load_frag(vp, vp + 16), o[j]);
    }
    __syncthreads();
  }

  float inv[8];
  #pragma unroll
  for (int r = 0; r < 8; ++r) inv[r] = 1.0f / rL[r];
  #pragma unroll
  for (int j = 0; j < 8; ++j)
    #pragma unroll
    for (int r = 0; r < 8; ++r) {
      const int m = qrow0 + r + 8*half;
      const int d = 16*j + lr;
      out[((size_t)(b*L + m) * NHEADS + h) * HDIM + d] = f2bf(o[j][r] * inv[r]);
    }
}

// ---------- host ----------
extern "C" void kernel_launch(void* const* d_in, const int* in_sizes, int n_in,
                              void* d_out, int out_size, void* d_ws, size_t ws_size,
                              hipStream_t stream) {
  const float* x        = (const float*)d_in[0];
  const int*   seq_lens = (const int*)  d_in[1];
  const int*   gsizes   = (const int*)  d_in[2];
  const float* freqs    = (const float*)d_in[3];
  const float* Wq = (const float*)d_in[4];  const float* bq = (const float*)d_in[5];
  const float* Wk = (const float*)d_in[6];  const float* bk = (const float*)d_in[7];
  const float* Wv = (const float*)d_in[8];  const float* bv = (const float*)d_in[9];
  const float* Wo = (const float*)d_in[10]; const float* bo = (const float*)d_in[11];
  const float* gq = (const float*)d_in[12]; const float* gk = (const float*)d_in[13];

  const size_t NTOK = (size_t)MROWS * DIM;     // 9,437,184
  const size_t WELT = (size_t)DIM * DIM;       // 2,359,296

  u16*   xb  = (u16*)d_ws;
  u16*   WtQ = xb  + NTOK;
  u16*   WtK = WtQ + WELT;
  u16*   WtV = WtK + WELT;
  u16*   WtO = WtV + WELT;
  float* qf  = (float*)(WtO + WELT);
  float* kf  = qf + NTOK;
  float* vf  = kf + NTOK;
  u16*   qb2 = (u16*)(vf + NTOK);
  u16*   kb2 = qb2 + NTOK;
  u16*   vb2 = kb2 + NTOK;
  u16*   aob = (u16*)qf;                       // alias: qf dead after norm_rope

  // prep: bf16 activations + transposed bf16 weights
  cvt_bf16<<<NTOK / 1024, 256, 0, stream>>>(x, xb);
  dim3 gt(DIM / 32, DIM / 32);                 // 48 x 48
  transpose_bf16<<<gt, 256, 0, stream>>>(Wq, WtQ, DIM, DIM);
  transpose_bf16<<<gt, 256, 0, stream>>>(Wk, WtK, DIM, DIM);
  transpose_bf16<<<gt, 256, 0, stream>>>(Wv, WtV, DIM, DIM);
  transpose_bf16<<<gt, 256, 0, stream>>>(Wo, WtO, DIM, DIM);

  dim3 gg(DIM / 128, MROWS / 128);             // 12 x 48
  gemm_bf16<<<gg, 256, 0, stream>>>(xb, WtQ, bq, qf, MROWS, DIM, DIM);
  gemm_bf16<<<gg, 256, 0, stream>>>(xb, WtK, bk, kf, MROWS, DIM, DIM);
  gemm_bf16<<<gg, 256, 0, stream>>>(xb, WtV, bv, vf, MROWS, DIM, DIM);

  norm_rope<<<dim3(MROWS, 3), 256, 0, stream>>>(qf, kf, vf, gq, gk, gsizes,
                                                freqs, qb2, kb2, vb2, SEQ);

  attn<<<dim3(SEQ / 128, NHEADS, BATCH), 256, 0, stream>>>(qb2, kb2, vb2,
                                                           seq_lens, aob, SEQ);

  gemm_bf16<<<gg, 256, 0, stream>>>(aob, WtO, bo, (float*)d_out, MROWS, DIM, DIM);
}
